// BEVDeformCrossAttn_53334903882108
// MI455X (gfx1250) — compile-verified, matrix operand order checked by reading, not run on hardware
//
#include <hip/hip_runtime.h>
#include <hip/hip_bf16.h>

// ---------------- problem constants ----------------
#define KD   6
#define BB   4
#define TT   12
#define QQ   72          // KD*TT
#define BQ   288         // BB*QQ
#define DIMC 256
#define HEADS_ 8
#define PTS_ 6
#define HD_  32
#define NPTS 13824       // BQ*HEADS_*PTS_
#define HF   200
#define WF   200
#define GS_  51.2f
#define SCALE_ 0.125f    // 64^-0.5

typedef _Float16 h8v  __attribute__((ext_vector_type(8)));
typedef _Float16 v16h __attribute__((ext_vector_type(16)));
typedef float    v8f  __attribute__((ext_vector_type(8)));

// ---------------- generic f16 WMMA GEMM: C = act(A @ B + bias) ----------------
// A: (M x Kd) f16 row-major.  Bt: (N x Kd) f16 row-major (i.e. B transposed).
// One wave per 16x16 output tile; K stepped by 32 with v_wmma_f32_16x16x32_f16.
__global__ __launch_bounds__(32) void gemm16(const _Float16* __restrict__ A,
                                             const _Float16* __restrict__ Bt,
                                             const float* __restrict__ bias,
                                             float* __restrict__ Cf,
                                             _Float16* __restrict__ Ch,
                                             int M, int N, int Kd, int relu)
{
  const int lane = threadIdx.x & 31;
  const int tm = blockIdx.x * 16;
  const int tn = blockIdx.y * 16;
  const int hs  = lane >> 4;     // half-wave select (ISA 16-bit A/B layout)
  const int r16 = lane & 15;
  const _Float16* ap = A  + (size_t)(tm + r16) * Kd + hs * 8;
  const _Float16* bp = Bt + (size_t)(tn + r16) * Kd + hs * 8;

  v8f acc = {0.f,0.f,0.f,0.f,0.f,0.f,0.f,0.f};
  for (int k0 = 0; k0 < Kd; k0 += 32) {
    h8v a0 = *(const h8v*)(ap + k0);        // K = c .. c+7   (c = 8*hs)
    h8v a1 = *(const h8v*)(ap + k0 + 16);   // K = c+16 .. c+23
    h8v b0 = *(const h8v*)(bp + k0);
    h8v b1 = *(const h8v*)(bp + k0 + 16);
    v16h a, b;
#pragma unroll
    for (int i = 0; i < 8; i++) { a[i] = a0[i]; a[i+8] = a1[i]; b[i] = b0[i]; b[i+8] = b1[i]; }
    acc = __builtin_amdgcn_wmma_f32_16x16x32_f16(false, a, false, b, (short)0, acc, false, false);
  }

  const int col = tn + r16;
  const float bv = bias ? bias[col] : 0.0f;
#pragma unroll
  for (int r = 0; r < 8; r++) {             // C/D layout: VGPR r -> row r + 8*hs
    int row = tm + hs * 8 + r;
    float v = acc[r] + bv;
    if (relu) v = fmaxf(v, 0.0f);
    if (Cf) Cf[(size_t)row * N + col] = v;
    if (Ch) Ch[(size_t)row * N + col] = (_Float16)v;
  }
}

// ---------------- small prep kernels ----------------
__global__ void cast_f16(const float* __restrict__ s, _Float16* __restrict__ d, int n)
{
  int i = blockIdx.x * blockDim.x + threadIdx.x;
  if (i < n) d[i] = (_Float16)s[i];
}

// Wt[n][k] = W[k][n], cast to f16 (coalesced writes along k)
__global__ void transpose_cast(const float* __restrict__ W, _Float16* __restrict__ Wt,
                               int Kd, int Nd)
{
  int i = blockIdx.x * blockDim.x + threadIdx.x;
  if (i >= Kd * Nd) return;
  int n = i / Kd, k = i % Kd;
  Wt[i] = (_Float16)W[(size_t)k * Nd + n];
}

// dec_embed (K,B,T,D) -> act (B*Q x D) f16, row m = b*QQ + k*TT + t
__global__ void cast_act(const float* __restrict__ de, _Float16* __restrict__ act)
{
  int i = blockIdx.x * blockDim.x + threadIdx.x;
  if (i >= BQ * DIMC) return;
  int d = i & 255, m = i >> 8;
  int b = m / QQ, q = m % QQ, k = q / TT, t = q % TT;
  act[i] = (_Float16)de[(((size_t)k * BB + b) * TT + t) * DIMC + d];
}

// ref_points (K,B,T,2) -> rpos (BQ x 2) in m-order
__global__ void reorder_rp(const float* __restrict__ rp, float* __restrict__ rpos)
{
  int i = blockIdx.x * blockDim.x + threadIdx.x;
  if (i >= BQ * 2) return;
  int c = i & 1, m = i >> 1;
  int b = m / QQ, q = m % QQ, k = q / TT, t = q % TT;
  rpos[i] = rp[(((size_t)k * BB + b) * TT + t) * 2 + c];
}

// ---------------- offset MLP (32-wide, exact GELU) + grid generation ----------------
__global__ void offsets_kernel(const float* __restrict__ conq, const float* __restrict__ rpos,
                               const float* __restrict__ w1, const float* __restrict__ b1,
                               const float* __restrict__ w2, const float* __restrict__ b2,
                               float* __restrict__ grid, float* __restrict__ vgq)
{
  int u = blockIdx.x * blockDim.x + threadIdx.x;   // u = m*H + h
  if (u >= BQ * HEADS_) return;
  int h = u % HEADS_, m = u / HEADS_;
  const float* cq = conq + (size_t)m * DIMC + h * HD_;
  float hb[HD_];
#pragma unroll
  for (int j = 0; j < HD_; j++) {
    float s = b1[j];
#pragma unroll
    for (int c = 0; c < HD_; c++) s += cq[c] * w1[c * HD_ + j];
    hb[j] = 0.5f * s * (1.0f + erff(s * 0.70710678118f));   // exact GELU
  }
  float rx = rpos[m * 2 + 0], ry = rpos[m * 2 + 1];
#pragma unroll
  for (int p = 0; p < PTS_; p++) {
    float ox = b2[2 * p], oy = b2[2 * p + 1];
#pragma unroll
    for (int c = 0; c < HD_; c++) {
      ox += hb[c] * w2[c * (2 * PTS_) + 2 * p];
      oy += hb[c] * w2[c * (2 * PTS_) + 2 * p + 1];
    }
    float vx = rx + ox, vy = ry + oy;                 // vgrid_q (metric)
    int n = u * PTS_ + p;
    vgq[2 * n] = vx; vgq[2 * n + 1] = vy;
    grid[2 * n]     = vx * (1.0f / GS_);              // x
    grid[2 * n + 1] = -(vy * (1.0f / GS_));           // y flipped
  }
}

// ---------------- DAB-DETR sine embedding: out[n] = cat(emb(y), emb(x)), 256-d ----------------
__global__ void sine_kernel(const float* __restrict__ pos, _Float16* __restrict__ out, int Npos)
{
  int i = blockIdx.x * blockDim.x + threadIdx.x;
  if (i >= Npos * 128) return;
  int n = i >> 7, d = i & 127;
  int half = d >> 6, j = d & 63;                       // half 0 -> y, 1 -> x
  float p = pos[n * 2 + (half ? 0 : 1)];
  float inv = __expf(-(float)j * (9.210340371976184f / 64.0f)); // 10000^(-j/64)
  float ang = p * 6.283185307179586f * inv;
  float s, c;
  __sincosf(ang, &s, &c);
  size_t base = (size_t)n * 256 + half * 128 + 2 * j;
  out[base] = (_Float16)s;
  out[base + 1] = (_Float16)c;
}

// ---------------- bilinear gather of raw 256-ch BEV feature at each grid point ----------------
// bev_feat stays in native (B,C,H,W) f32: 164 MB < 192 MB L2, scattered 4B reads are L2 hits.
__global__ __launch_bounds__(256) void gather_kernel(const float* __restrict__ bev,
                                                     const float* __restrict__ grid,
                                                     _Float16* __restrict__ sampled)
{
  int n = blockIdx.x;            // point index ((b*QQ+q)*H + h)*P + p
  int c = threadIdx.x;           // channel
  int b = n / (QQ * HEADS_ * PTS_);
  float gx = grid[2 * n], gy = grid[2 * n + 1];
  float x = (gx + 1.0f) * (WF * 0.5f) - 0.5f;
  float y = (gy + 1.0f) * (HF * 0.5f) - 0.5f;
  float xf = floorf(x), yf = floorf(y);
  int x0 = (int)xf, y0 = (int)yf;
  float wx1 = x - xf, wx0 = 1.0f - wx1;
  float wy1 = y - yf, wy0 = 1.0f - wy1;
  const float* base = bev + ((size_t)b * DIMC + c) * (HF * WF);
  float acc = 0.0f;
  auto tap = [&](int xi, int yi, float w) {
    if (xi >= 0 && xi < WF && yi >= 0 && yi < HF) acc += w * base[yi * WF + xi];
  };
  tap(x0,     y0,     wx0 * wy0);
  tap(x0 + 1, y0,     wx1 * wy0);
  tap(x0,     y0 + 1, wx0 * wy1);
  tap(x0 + 1, y0 + 1, wx1 * wy1);
  sampled[(size_t)n * DIMC + c] = (_Float16)acc;
}

// ---------------- attention over P=6 points, one wave per (m,h) ----------------
__global__ __launch_bounds__(32) void attn_kernel(const float* __restrict__ conq,
                                                  const float* __restrict__ posq,
                                                  const float* __restrict__ qs,
                                                  const _Float16* __restrict__ conk,
                                                  const float* __restrict__ posk,
                                                  const _Float16* __restrict__ vf,
                                                  _Float16* __restrict__ out)
{
  int u = blockIdx.x;                  // u = m*H + h
  int lane = threadIdx.x & 31;
  int h = u % HEADS_, m = u / HEADS_;
  int q = m % QQ, b = m / QQ, k_ = q / TT, t = q % TT;
  int col = h * HD_ + lane;
  float qc = conq[(size_t)m * DIMC + col];
  float sc = qs[(((size_t)k_ * BB + b) * TT + t) * DIMC + col];
  float qp = posq[(size_t)m * DIMC + col] * sc;
  float sim[PTS_];
#pragma unroll
  for (int p = 0; p < PTS_; p++) {
    size_t n = (size_t)u * PTS_ + p;
    float s = qc * (float)conk[n * DIMC + col] + qp * posk[n * HD_ + lane];
#pragma unroll
    for (int off = 16; off > 0; off >>= 1) s += __shfl_xor(s, off, 32);
    sim[p] = s * SCALE_;
  }
  float mx = sim[0];
#pragma unroll
  for (int p = 1; p < PTS_; p++) mx = fmaxf(mx, sim[p]);
  float e[PTS_]; float den = 0.0f;
#pragma unroll
  for (int p = 0; p < PTS_; p++) { e[p] = __expf(sim[p] - mx); den += e[p]; }
  float inv = 1.0f / den, o = 0.0f;
#pragma unroll
  for (int p = 0; p < PTS_; p++)
    o += e[p] * inv * (float)vf[((size_t)u * PTS_ + p) * DIMC + col];
  out[(size_t)m * DIMC + col] = (_Float16)o;
}

// ---------------- residual + layout restore ----------------
__global__ void finalize_kernel(const float* __restrict__ de, const float* __restrict__ proj,
                                float* __restrict__ out)
{
  int i = blockIdx.x * blockDim.x + threadIdx.x;
  if (i >= KD * BB * TT * DIMC) return;
  int d = i & 255, r = i >> 8;
  int t = r % TT; int r2 = r / TT; int b = r2 % BB; int k = r2 / BB;
  int m = b * QQ + k * TT + t;
  out[i] = de[i] + proj[(size_t)m * DIMC + d];
}

// ---------------- host orchestration ----------------
extern "C" void kernel_launch(void* const* d_in, const int* in_sizes, int n_in,
                              void* d_out, int out_size, void* d_ws, size_t ws_size,
                              hipStream_t stream)
{
  (void)in_sizes; (void)n_in; (void)out_size; (void)ws_size;
  const float* dec_embed = (const float*)d_in[0];
  const float* bev       = (const float*)d_in[1];
  const float* qs        = (const float*)d_in[2];
  const float* rp        = (const float*)d_in[3];
  const float* w_q = (const float*)d_in[4];  const float* b_q = (const float*)d_in[5];
  const float* w_k = (const float*)d_in[6];
  const float* w_v = (const float*)d_in[7];
  const float* pq_w1 = (const float*)d_in[8];  const float* pq_b1 = (const float*)d_in[9];
  const float* pq_w2 = (const float*)d_in[10]; const float* pq_b2 = (const float*)d_in[11];
  const float* pk_w1 = (const float*)d_in[12]; const float* pk_b1 = (const float*)d_in[13];
  const float* pk_w2 = (const float*)d_in[14]; const float* pk_b2 = (const float*)d_in[15];
  const float* off_w1 = (const float*)d_in[16]; const float* off_b1 = (const float*)d_in[17];
  const float* off_w2 = (const float*)d_in[18]; const float* off_b2 = (const float*)d_in[19];
  const float* w_o = (const float*)d_in[20]; const float* b_o = (const float*)d_in[21];
  float* out = (float*)d_out;

  char* cur = (char*)d_ws;
  auto alloc = [&](size_t bytes) -> void* {
    void* r = (void*)cur; cur += (bytes + 255) & ~(size_t)255; return r;
  };
  _Float16* actF   = (_Float16*)alloc((size_t)BQ * DIMC * 2);
  _Float16* wqT    = (_Float16*)alloc((size_t)256 * 256 * 2);
  _Float16* pqw1T  = (_Float16*)alloc((size_t)256 * 256 * 2);
  _Float16* pqw2T  = (_Float16*)alloc((size_t)256 * 256 * 2);
  _Float16* pkw1T  = (_Float16*)alloc((size_t)256 * 256 * 2);
  _Float16* pkw2T  = (_Float16*)alloc((size_t)32 * 256 * 2);
  _Float16* woT    = (_Float16*)alloc((size_t)256 * 256 * 2);
  _Float16* wkH    = (_Float16*)alloc((size_t)256 * 256 * 2);
  _Float16* wvH    = (_Float16*)alloc((size_t)256 * 256 * 2);
  float*    rpos   = (float*)alloc((size_t)BQ * 2 * 4);
  float*    conqF  = (float*)alloc((size_t)BQ * DIMC * 4);
  float*    gridB  = (float*)alloc((size_t)NPTS * 2 * 4);
  float*    vgqB   = (float*)alloc((size_t)NPTS * 2 * 4);
  _Float16* sineq  = (_Float16*)alloc((size_t)BQ * DIMC * 2);
  _Float16* posqh  = (_Float16*)alloc((size_t)BQ * DIMC * 2);
  float*    posqF  = (float*)alloc((size_t)BQ * DIMC * 4);
  _Float16* sinek  = (_Float16*)alloc((size_t)NPTS * DIMC * 2);
  _Float16* poskh  = (_Float16*)alloc((size_t)NPTS * DIMC * 2);
  float*    poskF  = (float*)alloc((size_t)NPTS * HD_ * 4);
  _Float16* sampled= (_Float16*)alloc((size_t)NPTS * DIMC * 2);
  _Float16* conkH  = (_Float16*)alloc((size_t)NPTS * DIMC * 2);
  _Float16* vH     = (_Float16*)alloc((size_t)NPTS * DIMC * 2);
  _Float16* attnH  = (_Float16*)alloc((size_t)BQ * DIMC * 2);
  float*    projF  = (float*)alloc((size_t)BQ * DIMC * 4);

  const int TPB = 256;
  // ---- prep: casts / transposes ----
  cast_act<<<(BQ * DIMC + TPB - 1) / TPB, TPB, 0, stream>>>(dec_embed, actF);
  transpose_cast<<<(65536 + TPB - 1) / TPB, TPB, 0, stream>>>(w_q,   wqT,   256, 256);
  transpose_cast<<<(65536 + TPB - 1) / TPB, TPB, 0, stream>>>(pq_w1, pqw1T, 256, 256);
  transpose_cast<<<(65536 + TPB - 1) / TPB, TPB, 0, stream>>>(pq_w2, pqw2T, 256, 256);
  transpose_cast<<<(65536 + TPB - 1) / TPB, TPB, 0, stream>>>(pk_w1, pkw1T, 256, 256);
  transpose_cast<<<(8192  + TPB - 1) / TPB, TPB, 0, stream>>>(pk_w2, pkw2T, 256, 32);
  transpose_cast<<<(65536 + TPB - 1) / TPB, TPB, 0, stream>>>(w_o,   woT,   256, 256);
  cast_f16<<<(65536 + TPB - 1) / TPB, TPB, 0, stream>>>(w_k, wkH, 65536); // already (N,K)
  cast_f16<<<(65536 + TPB - 1) / TPB, TPB, 0, stream>>>(w_v, wvH, 65536);
  reorder_rp<<<(BQ * 2 + TPB - 1) / TPB, TPB, 0, stream>>>(rp, rpos);

  // ---- con_q = dec @ w_q + b_q (WMMA) ----
  gemm16<<<dim3(BQ / 16, 256 / 16), 32, 0, stream>>>(actF, wqT, b_q, conqF, nullptr, BQ, 256, 256, 0);

  // ---- per-head offset MLP -> sampling grid + vgrid_q ----
  offsets_kernel<<<(BQ * HEADS_ + 127) / 128, 128, 0, stream>>>(conqF, rpos, off_w1, off_b1,
                                                                off_w2, off_b2, gridB, vgqB);

  // ---- sine embeddings ----
  sine_kernel<<<((size_t)BQ * 128 + TPB - 1) / TPB, TPB, 0, stream>>>(rpos, sineq, BQ);
  sine_kernel<<<((size_t)NPTS * 128 + TPB - 1) / TPB, TPB, 0, stream>>>(vgqB, sinek, NPTS);

  // ---- bilinear gather in raw 256-ch space (sample-then-project rewrite) ----
  gather_kernel<<<NPTS, 256, 0, stream>>>(bev, gridB, sampled);

  // ---- pos_q MLP (WMMA) ----
  gemm16<<<dim3(BQ / 16, 16), 32, 0, stream>>>(sineq, pqw1T, pq_b1, nullptr, posqh, BQ, 256, 256, 1);
  gemm16<<<dim3(BQ / 16, 16), 32, 0, stream>>>(posqh, pqw2T, pq_b2, posqF, nullptr, BQ, 256, 256, 0);

  // ---- pos_k MLP (WMMA) ----
  gemm16<<<dim3(NPTS / 16, 16), 32, 0, stream>>>(sinek, pkw1T, pk_b1, nullptr, poskh, NPTS, 256, 256, 1);
  gemm16<<<dim3(NPTS / 16, 2),  32, 0, stream>>>(poskh, pkw2T, pk_b2, poskF, nullptr, NPTS, 32, 256, 0);

  // ---- deferred 1x1 conv: project sampled features with w_k / w_v (WMMA) ----
  gemm16<<<dim3(NPTS / 16, 16), 32, 0, stream>>>(sampled, wkH, nullptr, nullptr, conkH, NPTS, 256, 256, 0);
  gemm16<<<dim3(NPTS / 16, 16), 32, 0, stream>>>(sampled, wvH, nullptr, nullptr, vH,    NPTS, 256, 256, 0);

  // ---- softmax attention over 6 points ----
  attn_kernel<<<BQ * HEADS_, 32, 0, stream>>>(conqF, posqF, qs, conkH, poskF, vH, attnH);

  // ---- output projection + residual ----
  gemm16<<<dim3(BQ / 16, 16), 32, 0, stream>>>(attnH, woT, b_o, projF, nullptr, BQ, 256, 256, 0);
  finalize_kernel<<<(KD * BB * TT * DIMC + TPB - 1) / TPB, TPB, 0, stream>>>(dec_embed, projF, out);
}
